// NLSGCRNCNNCell_57939108823839
// MI455X (gfx1250) — matrix operand order repaired and checked by reading, not compiled, not run on hardware
//
#include <hip/hip_runtime.h>
#include <cstdint>
#include <cstddef>

// ---------------------------------------------------------------------------
// Types for CDNA5 WMMA (wave32, v_wmma_f32_16x16x32_bf16)
// ---------------------------------------------------------------------------
typedef __bf16 bf16_t;
typedef __attribute__((ext_vector_type(16))) __bf16 v16bf;
typedef __attribute__((ext_vector_type(8)))  float  v8f;
typedef int v4i_t __attribute__((vector_size(16)));   // matches builtin param type

union V16 { v16bf v; uint4 q[2]; };
static_assert(sizeof(V16) == 32, "V16 size");

__device__ __forceinline__ v8f wmma_bf16(v16bf a, v16bf b, v8f c) {
  // 8 args: (neg_a, A, neg_b, B, c_mod, C, reuse_a, reuse_b)
  return __builtin_amdgcn_wmma_f32_16x16x32_bf16(false, a, false, b, (short)0, c, false, false);
}
__device__ __forceinline__ v8f v8zero() {
  v8f v = {0.f,0.f,0.f,0.f,0.f,0.f,0.f,0.f};
  return v;
}
__device__ __forceinline__ uint4 ld16(const bf16_t* p) { return *(const uint4*)p; }

// --------------------------- async global->LDS -----------------------------
#if defined(__has_builtin)
#  if __has_builtin(__builtin_amdgcn_global_load_async_to_lds_b128)
#    define HAVE_ASYNC_LDS 1
#  endif
#endif
#ifndef HAVE_ASYNC_LDS
#  define HAVE_ASYNC_LDS 0
#endif

__device__ __forceinline__ void cp16_to_lds(const bf16_t* g, bf16_t* l) {
#if HAVE_ASYNC_LDS
  __builtin_amdgcn_global_load_async_to_lds_b128((v4i_t*)g, (v4i_t*)l, 0, 0);
#else
  *(uint4*)l = *(const uint4*)g;   // global_load_b128 + ds_store_b128
#endif
}
__device__ __forceinline__ void wait_async_lds() {
#if HAVE_ASYNC_LDS
#  if __has_builtin(__builtin_amdgcn_s_wait_asynccnt)
  __builtin_amdgcn_s_wait_asynccnt(0);
#  else
  asm volatile("s_wait_asynccnt 0" ::: "memory");
#  endif
#endif
}

#define NN 2000   // nodes
#define NP 2016   // nodes padded to 63*32 (K-step multiple); 2016 = 21*96
#define BB 32     // batch

// ---------------------------------------------------------------------------
// Utility kernels
// ---------------------------------------------------------------------------
__global__ void k_fill(uint32_t* __restrict__ p, size_t n) {
  size_t stride = (size_t)gridDim.x * blockDim.x;
  for (size_t i = (size_t)blockIdx.x * blockDim.x + threadIdx.x; i < n; i += stride) p[i] = 0u;
}

// emb (2000x16 f32) -> emb_bf (2000x32 bf16, d>=16 zero)
__global__ void k_embprep(const float* __restrict__ emb, bf16_t* __restrict__ out) {
  int i = blockIdx.x * 256 + threadIdx.x;
  if (i >= NN * 32) return;
  int d = i & 31, n = i >> 5;
  out[i] = (d < 16) ? (bf16_t)emb[n * 16 + d] : (bf16_t)0.f;
}

// wp (16 x J f32, d-major) -> wpT (J x 32 bf16, d>=16 zero): B-operand for weights GEMM
__global__ void k_wprep(const float* __restrict__ src, bf16_t* __restrict__ dst, int J) {
  int j = blockIdx.x * 256 + threadIdx.x;
  if (j >= J) return;
  for (int d = 0; d < 16; ++d) dst[(size_t)j * 32 + d] = (bf16_t)src[(size_t)d * J + j];
  for (int d = 16; d < 32; ++d) dst[(size_t)j * 32 + d] = (bf16_t)0.f;
}

// bias[n][o] = sum_d emb[n][d] * bp[d][o]
__global__ void k_bias(const float* __restrict__ emb, const float* __restrict__ bp,
                       float* __restrict__ out, int O) {
  int i = blockIdx.x * 256 + threadIdx.x;
  if (i >= NN * O) return;
  int o = i % O, n = i / O;
  float a = 0.f;
  for (int d = 0; d < 16; ++d) a += emb[n * 16 + d] * bp[d * O + o];
  out[i] = a;
}

// s = softmax(relu(emb @ emb^T)) ; write s_bf (row major, NP-padded) and sT_bf.
__global__ __launch_bounds__(256) void k_softmax(const float* __restrict__ emb,
                                                 bf16_t* __restrict__ s, bf16_t* __restrict__ sT) {
  __shared__ float row[NP];
  __shared__ float red[256];
  int n = blockIdx.x, t = threadIdx.x;
  float e[16];
  if (n < NN) {
    #pragma unroll
    for (int d = 0; d < 16; ++d) e[d] = emb[n * 16 + d];
  }
  float lmax = 0.f;  // relu >= 0
  for (int m = t; m < NP; m += 256) {
    float v = 0.f;
    if (n < NN && m < NN) {
      float acc = 0.f;
      #pragma unroll
      for (int d = 0; d < 16; ++d) acc += e[d] * emb[m * 16 + d];
      v = acc > 0.f ? acc : 0.f;
    }
    row[m] = v;
    lmax = fmaxf(lmax, v);
  }
  red[t] = lmax; __syncthreads();
  for (int s2 = 128; s2 > 0; s2 >>= 1) { if (t < s2) red[t] = fmaxf(red[t], red[t + s2]); __syncthreads(); }
  float mx = red[0]; __syncthreads();
  float lsum = 0.f;
  for (int m = t; m < NP; m += 256) {
    float p = (n < NN && m < NN) ? __expf(row[m] - mx) : 0.f;
    row[m] = p; lsum += p;
  }
  red[t] = lsum; __syncthreads();
  for (int s2 = 128; s2 > 0; s2 >>= 1) { if (t < s2) red[t] += red[t + s2]; __syncthreads(); }
  float inv = (n < NN) ? 1.f / red[0] : 0.f;
  for (int m = t; m < NP; m += 256) {
    bf16_t pb = (bf16_t)(row[m] * inv);
    s[(size_t)n * NP + m] = pb;
    sT[(size_t)m * NP + n] = pb;
  }
}

// ---------------------------------------------------------------------------
// s2 = s @ s  (A = s row-major, B via sT; 32 rows x 96 cols per wave)
// ---------------------------------------------------------------------------
__global__ __launch_bounds__(32) void k_s2(const bf16_t* __restrict__ S,
                                           const bf16_t* __restrict__ ST,
                                           bf16_t* __restrict__ S2) {
  int m0 = blockIdx.x * 32, c0 = blockIdx.y * 96;
  int L = threadIdx.x, lm = L & 15, lh = L >> 4;
  v8f acc[2][6];
  #pragma unroll
  for (int mt = 0; mt < 2; ++mt)
    #pragma unroll
    for (int ct = 0; ct < 6; ++ct) acc[mt][ct] = v8zero();
  const bf16_t* ar0base = S + (size_t)(m0 + lm) * NP + lh * 8;
  for (int ks = 0; ks < NP / 32; ++ks) {
    int kb = ks * 32;
    V16 a0, a1;
    const bf16_t* ar0 = ar0base + kb;
    a0.q[0] = ld16(ar0);      a0.q[1] = ld16(ar0 + 16);
    const bf16_t* ar1 = ar0 + (size_t)16 * NP;
    a1.q[0] = ld16(ar1);      a1.q[1] = ld16(ar1 + 16);
    #pragma unroll
    for (int ct = 0; ct < 6; ++ct) {
      int col = c0 + ct * 16 + lm;
      const bf16_t* br = ST + (size_t)col * NP + kb + lh * 8;
      V16 bv; bv.q[0] = ld16(br); bv.q[1] = ld16(br + 16);
      acc[0][ct] = wmma_bf16(a0.v, bv.v, acc[0][ct]);
      acc[1][ct] = wmma_bf16(a1.v, bv.v, acc[1][ct]);
    }
  }
  #pragma unroll
  for (int mt = 0; mt < 2; ++mt)
    #pragma unroll
    for (int ct = 0; ct < 6; ++ct) {
      int col = c0 + ct * 16 + lm;
      #pragma unroll
      for (int r = 0; r < 8; ++r) {
        int n = m0 + mt * 16 + r + lh * 8;
        S2[(size_t)n * NP + col] = (bf16_t)acc[mt][ct][r];
      }
    }
}

// ---------------------------------------------------------------------------
// Per-node weight pools: Wt[n][o][ki] = sum_d emb[n][d]*wp[d][ki][o]
// GEMM (2000 x 32pad) @ (32pad x J), output stored transposed per node.
// ---------------------------------------------------------------------------
__global__ __launch_bounds__(32) void k_wgemm(const bf16_t* __restrict__ A,
                                              const bf16_t* __restrict__ BT,
                                              bf16_t* __restrict__ out, int O, int KP) {
  int m0 = blockIdx.x * 16, j0 = blockIdx.y * 16;
  int L = threadIdx.x, lm = L & 15, lh = L >> 4;
  V16 a, b;
  const bf16_t* ar = A + (size_t)(m0 + lm) * 32 + lh * 8;
  a.q[0] = ld16(ar);      a.q[1] = ld16(ar + 16);
  const bf16_t* br = BT + (size_t)(j0 + lm) * 32 + lh * 8;
  b.q[0] = ld16(br);      b.q[1] = ld16(br + 16);
  v8f acc = wmma_bf16(a.v, b.v, v8zero());
  int j = j0 + lm, o = j % O, ki = j / O;
  #pragma unroll
  for (int r = 0; r < 8; ++r) {
    int n = m0 + r + lh * 8;
    out[(size_t)n * O * KP + (size_t)o * KP + ki] = (bf16_t)acc[r];
  }
}

// ---------------------------------------------------------------------------
// xT[b][n][c] = sum_t T[t]*x_full[b][t][n][c]  for g and u (c>=48 zero pad)
// ---------------------------------------------------------------------------
__global__ void k_xt(const float* __restrict__ xf, const float* __restrict__ gT,
                     const float* __restrict__ uT, bf16_t* __restrict__ xTg,
                     bf16_t* __restrict__ xTu) {
  long i = (long)blockIdx.x * 256 + threadIdx.x;
  if (i >= (long)BB * NN * 64) return;
  int c = (int)(i & 63);
  long nn = i >> 6;
  int n = (int)(nn % NN), b = (int)(nn / NN);
  float ag = 0.f, au = 0.f;
  if (c < 48) {
    #pragma unroll
    for (int t = 0; t < 12; ++t) {
      float v = xf[(((size_t)b * 12 + t) * NN + n) * 48 + c];
      ag += v * gT[t];
      au += v * uT[t];
    }
  }
  xTg[i] = (bf16_t)ag;
  xTu[i] = (bf16_t)au;
}

// ---------------------------------------------------------------------------
// Build XcT (B,96,NP) bf16 (columns contiguous for B-operand) and the k=0 slot
// of x_g. mode 0: Xc=[x | state] (c 0..95). mode 1: c 32..95 = z*state.
// ---------------------------------------------------------------------------
__global__ void k_xc(const float* __restrict__ x, const float* __restrict__ st,
                     const float* __restrict__ z, bf16_t* __restrict__ XcT,
                     bf16_t* __restrict__ xg, int mode) {
  int cs = mode ? 64 : 96;
  long total = (long)BB * cs * NP;
  long i = (long)blockIdx.x * 256 + threadIdx.x;
  if (i >= total) return;
  int n = (int)(i % NP);
  long t = i / NP;
  int cr = (int)(t % cs), b = (int)(t / cs);
  int c = mode ? 32 + cr : cr;
  float v = 0.f;
  if (n < NN) {
    if (c < 32) v = x[((size_t)b * NN + n) * 32 + c];
    else {
      float sv = st[((size_t)b * NN + n) * 64 + (c - 32)];
      v = mode ? z[((size_t)b * NN + n) * 64 + (c - 32)] * sv : sv;
    }
  }
  bf16_t bv = (bf16_t)v;
  XcT[((size_t)b * 96 + c) * NP + n] = bv;
  if (n < NN) xg[((size_t)b * NN + n) * 288 + c] = bv;   // k=0 slot
}

// ---------------------------------------------------------------------------
// Diffusion: x_g[b][n][k][c] = (supports[k] @ Xc[b])[n][c], k=1 (s), k=2 (s2).
// Block = 3 waves (96 rows); shared double-buffered LDS B-panel filled with
// async global->LDS copies (ASYNCcnt) and consumed via ds_load_b128.
// ---------------------------------------------------------------------------
template <int CSTART, int CT>
__global__ __launch_bounds__(96) void k_diff(const bf16_t* __restrict__ S1,
                                             const bf16_t* __restrict__ S2,
                                             const bf16_t* __restrict__ XcT,
                                             bf16_t* __restrict__ xg) {
  __shared__ bf16_t Bpanel[2][32 * CT * 16];   // [buf][c_local*32 + kk]
  int b = blockIdx.z >> 1;
  int k = (blockIdx.z & 1) + 1;
  const bf16_t* S = (blockIdx.z & 1) ? S2 : S1;
  int wave = threadIdx.x >> 5;
  int m0 = blockIdx.x * 96 + wave * 32;        // 21 blocks * 96 rows = 2016
  int L = threadIdx.x & 31, lm = L & 15, lh = L >> 4;
  const bf16_t* Bb = XcT + ((size_t)b * 96 + CSTART * 16) * NP;

  v8f acc[2][CT];
  #pragma unroll
  for (int mt = 0; mt < 2; ++mt)
    #pragma unroll
    for (int ct = 0; ct < CT; ++ct) acc[mt][ct] = v8zero();

  const bf16_t* ar0base = S + (size_t)(m0 + lm) * NP + lh * 8;

  // Cooperative B-panel loader: thread t owns rows c = t, t+96, ... (64B each)
  auto load_panel = [&](int ks, int buf) {
    int kb = ks * 32;
    for (int c = threadIdx.x; c < CT * 16; c += 96) {
      const bf16_t* g = Bb + (size_t)c * NP + kb;
      bf16_t* l = &Bpanel[buf][c * 32];
      #pragma unroll
      for (int p = 0; p < 4; ++p) cp16_to_lds(g + p * 8, l + p * 8);
    }
  };

  load_panel(0, 0);
  wait_async_lds();
  __syncthreads();

  for (int ks = 0; ks < NP / 32; ++ks) {
    int buf = ks & 1;
    if (ks + 1 < NP / 32) load_panel(ks + 1, buf ^ 1);

    int kb = ks * 32;
    V16 a0, a1;
    const bf16_t* ar0 = ar0base + kb;
    a0.q[0] = ld16(ar0);      a0.q[1] = ld16(ar0 + 16);
    const bf16_t* ar1 = ar0 + (size_t)16 * NP;
    a1.q[0] = ld16(ar1);      a1.q[1] = ld16(ar1 + 16);
    #pragma unroll
    for (int ct = 0; ct < CT; ++ct) {
      const bf16_t* lr = &Bpanel[buf][(ct * 16 + lm) * 32 + lh * 8];
      V16 bv;
      bv.q[0] = *(const uint4*)lr;        // ds_load_b128
      bv.q[1] = *(const uint4*)(lr + 16);
      acc[0][ct] = wmma_bf16(a0.v, bv.v, acc[0][ct]);
      acc[1][ct] = wmma_bf16(a1.v, bv.v, acc[1][ct]);
    }
    wait_async_lds();
    __syncthreads();
  }

  #pragma unroll
  for (int mt = 0; mt < 2; ++mt)
    #pragma unroll
    for (int ct = 0; ct < CT; ++ct) {
      int c = (CSTART + ct) * 16 + lm;
      #pragma unroll
      for (int r = 0; r < 8; ++r) {
        int n = m0 + mt * 16 + r + lh * 8;
        if (n < NN) xg[(((size_t)b * NN + n) * 3 + k) * 96 + c] = (bf16_t)acc[mt][ct][r];
      }
    }
}

// ---------------------------------------------------------------------------
// Per-node GEMM + GRU gate epilogue (g path): out cols 0..63 -> z, 64..127 -> r
// ---------------------------------------------------------------------------
__global__ __launch_bounds__(64) void k_node_g(const bf16_t* __restrict__ xg,
                                               const bf16_t* __restrict__ xT,
                                               const bf16_t* __restrict__ Wp,
                                               const bf16_t* __restrict__ Ww,
                                               const float* __restrict__ bias,
                                               float* __restrict__ zbuf,
                                               float* __restrict__ rbuf) {
  int n = blockIdx.x;
  int L = threadIdx.x & 31, wave = threadIdx.x >> 5;
  int lm = L & 15, lh = L >> 4;
  int m0 = wave * 16;
  const bf16_t* Ar1 = xg + ((size_t)(m0 + lm) * NN + n) * 288 + lh * 8;
  const bf16_t* Ar2 = xT + ((size_t)(m0 + lm) * NN + n) * 64 + lh * 8;
  #pragma unroll
  for (int ot = 0; ot < 8; ++ot) {
    v8f acc = v8zero();
    if (ot < 4) {
      int oc = ot * 16 + lm;
      const bf16_t* Br = Wp + ((size_t)n * 64 + oc) * 288 + lh * 8;
      #pragma unroll
      for (int ks = 0; ks < 9; ++ks) {
        int kb = ks * 32;
        V16 a, b;
        a.q[0] = ld16(Ar1 + kb); a.q[1] = ld16(Ar1 + kb + 16);
        b.q[0] = ld16(Br + kb);  b.q[1] = ld16(Br + kb + 16);
        acc = wmma_bf16(a.v, b.v, acc);
      }
    } else {
      int oc = (ot - 4) * 16 + lm;
      const bf16_t* Br = Ww + ((size_t)n * 64 + oc) * 64 + lh * 8;
      #pragma unroll
      for (int ks = 0; ks < 2; ++ks) {
        int kb = ks * 32;
        V16 a, b;
        a.q[0] = ld16(Ar2 + kb); a.q[1] = ld16(Ar2 + kb + 16);
        b.q[0] = ld16(Br + kb);  b.q[1] = ld16(Br + kb + 16);
        acc = wmma_bf16(a.v, b.v, acc);
      }
    }
    int col = ot * 16 + lm;
    float bvs = bias[(size_t)n * 128 + col];
    #pragma unroll
    for (int r = 0; r < 8; ++r) {
      int brow = m0 + r + lh * 8;
      float v = 1.f / (1.f + __expf(-(acc[r] + bvs)));
      if (ot < 4) zbuf[((size_t)brow * NN + n) * 64 + col] = v;
      else        rbuf[((size_t)brow * NN + n) * 64 + (col - 64)] = v;
    }
  }
}

// ---------------------------------------------------------------------------
// Per-node GEMM + GRU candidate epilogue (u path): h = r*state + (1-r)*tanh(.)
// ---------------------------------------------------------------------------
__global__ __launch_bounds__(64) void k_node_u(const bf16_t* __restrict__ xg,
                                               const bf16_t* __restrict__ xT,
                                               const bf16_t* __restrict__ Wp,
                                               const bf16_t* __restrict__ Ww,
                                               const float* __restrict__ bias,
                                               const float* __restrict__ rbuf,
                                               const float* __restrict__ st,
                                               float* __restrict__ out) {
  int n = blockIdx.x;
  int L = threadIdx.x & 31, wave = threadIdx.x >> 5;
  int lm = L & 15, lh = L >> 4;
  int m0 = wave * 16;
  const bf16_t* Ar1 = xg + ((size_t)(m0 + lm) * NN + n) * 288 + lh * 8;
  const bf16_t* Ar2 = xT + ((size_t)(m0 + lm) * NN + n) * 64 + lh * 8;
  #pragma unroll
  for (int ot = 0; ot < 4; ++ot) {
    v8f acc = v8zero();
    if (ot < 2) {
      int oc = ot * 16 + lm;
      const bf16_t* Br = Wp + ((size_t)n * 32 + oc) * 288 + lh * 8;
      #pragma unroll
      for (int ks = 0; ks < 9; ++ks) {
        int kb = ks * 32;
        V16 a, b;
        a.q[0] = ld16(Ar1 + kb); a.q[1] = ld16(Ar1 + kb + 16);
        b.q[0] = ld16(Br + kb);  b.q[1] = ld16(Br + kb + 16);
        acc = wmma_bf16(a.v, b.v, acc);
      }
    } else {
      int oc = (ot - 2) * 16 + lm;
      const bf16_t* Br = Ww + ((size_t)n * 32 + oc) * 64 + lh * 8;
      #pragma unroll
      for (int ks = 0; ks < 2; ++ks) {
        int kb = ks * 32;
        V16 a, b;
        a.q[0] = ld16(Ar2 + kb); a.q[1] = ld16(Ar2 + kb + 16);
        b.q[0] = ld16(Br + kb);  b.q[1] = ld16(Br + kb + 16);
        acc = wmma_bf16(a.v, b.v, acc);
      }
    }
    int col = ot * 16 + lm;
    float bvs = bias[(size_t)n * 64 + col];
    #pragma unroll
    for (int r = 0; r < 8; ++r) {
      int brow = m0 + r + lh * 8;
      size_t oidx = ((size_t)brow * NN + n) * 64 + col;
      float hc = tanhf(acc[r] + bvs);
      float rr = rbuf[oidx];
      float sv = st[oidx];
      out[oidx] = rr * sv + (1.f - rr) * hc;
    }
  }
}

// ---------------------------------------------------------------------------
// Launch
// ---------------------------------------------------------------------------
extern "C" void kernel_launch(void* const* d_in, const int* in_sizes, int n_in,
                              void* d_out, int out_size, void* d_ws, size_t ws_size,
                              hipStream_t stream) {
  const float* x       = (const float*)d_in[0];
  const float* state   = (const float*)d_in[1];
  const float* x_full  = (const float*)d_in[2];
  const float* emb     = (const float*)d_in[3];
  // d_in[4] = zigzag_PI (unused by reference)
  const float* gw_pool = (const float*)d_in[5];
  const float* gw_win  = (const float*)d_in[6];
  const float* gb_pool = (const float*)d_in[7];
  const float* gT      = (const float*)d_in[8];
  const float* uw_pool = (const float*)d_in[9];
  const float* uw_win  = (const float*)d_in[10];
  const float* ub_pool = (const float*)d_in[11];
  const float* uT      = (const float*)d_in[12];
  float* out = (float*)d_out;

  char* w = (char*)d_ws;
  size_t off = 0;
  auto alloc = [&](size_t bytes) -> char* {
    char* p = w + off;
    off += (bytes + 255) & ~(size_t)255;
    return p;
  };
  bf16_t* s_bf   = (bf16_t*)alloc((size_t)NP * NP * 2);
  bf16_t* sT_bf  = (bf16_t*)alloc((size_t)NP * NP * 2);
  bf16_t* s2_bf  = (bf16_t*)alloc((size_t)NP * NP * 2);
  bf16_t* emb_bf = (bf16_t*)alloc((size_t)NN * 32 * 2);
  bf16_t* wpTg   = (bf16_t*)alloc((size_t)18432 * 32 * 2);
  bf16_t* wwTg   = (bf16_t*)alloc((size_t)3072 * 32 * 2);
  bf16_t* wpTu   = (bf16_t*)alloc((size_t)9216 * 32 * 2);
  bf16_t* wwTu   = (bf16_t*)alloc((size_t)1536 * 32 * 2);
  bf16_t* WgPool = (bf16_t*)alloc((size_t)NN * 64 * 288 * 2);
  bf16_t* WgWin  = (bf16_t*)alloc((size_t)NN * 64 * 64 * 2);
  bf16_t* WuPool = (bf16_t*)alloc((size_t)NN * 32 * 288 * 2);
  bf16_t* WuWin  = (bf16_t*)alloc((size_t)NN * 32 * 64 * 2);
  float*  biasg  = (float*) alloc((size_t)NN * 128 * 4);
  float*  biasu  = (float*) alloc((size_t)NN * 64 * 4);
  bf16_t* XcT    = (bf16_t*)alloc((size_t)BB * 96 * NP * 2);
  bf16_t* xg     = (bf16_t*)alloc((size_t)BB * NN * 288 * 2);
  bf16_t* xTg    = (bf16_t*)alloc((size_t)BB * NN * 64 * 2);
  bf16_t* xTu    = (bf16_t*)alloc((size_t)BB * NN * 64 * 2);
  float*  zbuf   = (float*) alloc((size_t)BB * NN * 64 * 4);
  float*  rbuf   = (float*) alloc((size_t)BB * NN * 64 * 4);

  auto cdiv = [](long a, long b) { return (int)((a + b - 1) / b); };

  // Zero the K-padded regions of the win-weight buffers (ki 48..63 never written)
  k_fill<<<2048, 256, 0, stream>>>((uint32_t*)WgWin, (size_t)NN * 64 * 64 * 2 / 4);
  k_fill<<<2048, 256, 0, stream>>>((uint32_t*)WuWin, (size_t)NN * 32 * 64 * 2 / 4);

  // Precompute (shared between both _tfl calls)
  k_embprep<<<cdiv((long)NN * 32, 256), 256, 0, stream>>>(emb, emb_bf);
  k_wprep<<<cdiv(18432, 256), 256, 0, stream>>>(gw_pool, wpTg, 18432);
  k_wprep<<<cdiv(3072, 256), 256, 0, stream>>>(gw_win, wwTg, 3072);
  k_wprep<<<cdiv(9216, 256), 256, 0, stream>>>(uw_pool, wpTu, 9216);
  k_wprep<<<cdiv(1536, 256), 256, 0, stream>>>(uw_win, wwTu, 1536);
  k_bias<<<cdiv((long)NN * 128, 256), 256, 0, stream>>>(emb, gb_pool, biasg, 128);
  k_bias<<<cdiv((long)NN * 64, 256), 256, 0, stream>>>(emb, ub_pool, biasu, 64);

  k_softmax<<<NP, 256, 0, stream>>>(emb, s_bf, sT_bf);
  k_s2<<<dim3(NP / 32, NP / 96), 32, 0, stream>>>(s_bf, sT_bf, s2_bf);

  k_wgemm<<<dim3(NN / 16, 18432 / 16), 32, 0, stream>>>(emb_bf, wpTg, WgPool, 64, 288);
  k_wgemm<<<dim3(NN / 16, 3072 / 16), 32, 0, stream>>>(emb_bf, wwTg, WgWin, 64, 64);
  k_wgemm<<<dim3(NN / 16, 9216 / 16), 32, 0, stream>>>(emb_bf, wpTu, WuPool, 32, 288);
  k_wgemm<<<dim3(NN / 16, 1536 / 16), 32, 0, stream>>>(emb_bf, wwTu, WuWin, 32, 64);

  k_xt<<<cdiv((long)BB * NN * 64, 256), 256, 0, stream>>>(x_full, gT, uT, xTg, xTu);

  // Phase 1: gate path (z, r)
  k_xc<<<cdiv((long)BB * 96 * NP, 256), 256, 0, stream>>>(x, state, zbuf, XcT, xg, 0);
  k_diff<0, 6><<<dim3(NP / 96, 1, BB * 2), 96, 0, stream>>>(s_bf, s2_bf, XcT, xg);
  k_node_g<<<NN, 64, 0, stream>>>(xg, xTg, WgPool, WgWin, biasg, zbuf, rbuf);

  // Phase 2: candidate path (only channels 32..95 change: z*state)
  k_xc<<<cdiv((long)BB * 64 * NP, 256), 256, 0, stream>>>(x, state, zbuf, XcT, xg, 1);
  k_diff<2, 4><<<dim3(NP / 96, 1, BB * 2), 96, 0, stream>>>(s_bf, s2_bf, XcT, xg);
  k_node_u<<<NN, 64, 0, stream>>>(xg, xTu, WuPool, WuWin, biasu, rbuf, state, out);
}
